// AttnHead_38852274160085
// MI455X (gfx1250) — matrix operand order repaired and checked by reading, not compiled, not run on hardware
//
#include <hip/hip_runtime.h>

// ---------------------------------------------------------------------------
// GAT attention head, fused flash-attention style for MI455X (gfx1250).
//   K1: seq_fts = seq @ W^T via V_WMMA_F32_16X16X4_F32 (f32 precision),
//       emits transposed f16 value matrix Vt[b][f][j] + f1/f2 row scores.
//   K2: per 16-row query block: stream bias_mat (536 MB = the HBM roofline
//       term, ~23us @ 23.3 TB/s) with non-temporal loads + prefetch, online
//       softmax, P@V via V_WMMA_F32_16X16X32_F16 with f32 accumulation.
//       No [N,N] matrix is ever materialized in memory.
// ---------------------------------------------------------------------------

typedef __attribute__((ext_vector_type(2)))  float    v2f;
typedef __attribute__((ext_vector_type(4)))  float    v4f;
typedef __attribute__((ext_vector_type(8)))  float    v8f;
typedef __attribute__((ext_vector_type(8)))  _Float16 v8h;
typedef __attribute__((ext_vector_type(16))) _Float16 v16h;

#define LOG2E 1.44269504088896340736f

static __device__ __forceinline__ float fexp2(float x) {
  return __builtin_amdgcn_exp2f(x);
}

// ===========================================================================
// Kernel 1: one wave per 16-row tile.
//   seq [B,N,128] f32  ->  Vt [B,64,N] f16 (transposed), f1/f2 [B,N] f32
// WMMA f32 16x16x4: A = seq rows (16x4), B = W^T (4x16), C = 16x16 f32.
// A layout: lanes 0-15 hold M=0..15 K={0,1}; lanes 16-31 hold K={2,3}.
// C layout: VGPR r -> row r + 8*hi, col = lane&15.
// ===========================================================================
__global__ __launch_bounds__(256) void proj_kernel(
    const float* __restrict__ seq, const float* __restrict__ W,
    const float* __restrict__ a1w, const float* __restrict__ a1b,
    const float* __restrict__ a2w, const float* __restrict__ a2b,
    _Float16* __restrict__ Vt, float* __restrict__ f1, float* __restrict__ f2,
    int N)
{
  const int lane = threadIdx.x & 31;
  const int wave = threadIdx.x >> 5;
  const int tile = blockIdx.x * 8 + wave;   // B*N/16 = 1024 tiles
  const int b    = tile >> 9;               // tile / (N/16), N = 8192
  const int i0   = (tile & 511) << 4;
  const int hi   = lane >> 4;
  const int nn   = lane & 15;

  const float* arow = seq + ((size_t)b * N + i0 + nn) * 128;

  v8f acc[4];
#pragma unroll
  for (int t = 0; t < 4; ++t) acc[t] = (v8f)0.f;

  for (int kk = 0; kk < 32; ++kk) {
    const int k0 = kk * 4 + 2 * hi;
    v2f a;
    a.x = arow[k0];
    a.y = arow[k0 + 1];
#pragma unroll
    for (int t = 0; t < 4; ++t) {
      const float* wrow = W + (size_t)(t * 16 + nn) * 128 + k0;  // W[f][k]
      v2f bv;
      bv.x = wrow[0];
      bv.y = wrow[1];
      acc[t] = __builtin_amdgcn_wmma_f32_16x16x4_f32(
          false, a, false, bv, (short)0, acc[t], false, false);
    }
  }

  // Store transposed f16 value matrix: Vt[b][f][j], 8 consecutive j per lane.
#pragma unroll
  for (int t = 0; t < 4; ++t) {
    v8h hv;
#pragma unroll
    for (int r = 0; r < 8; ++r) hv[r] = (_Float16)acc[t][r];
    *(v8h*)(Vt + ((size_t)b * 64 + t * 16 + nn) * N + i0 + 8 * hi) = hv;
  }

  // f1/f2: per-row dot with a1w/a2w; lane holds col nn, rows (8*hi + r).
  float p1[8], p2[8];
#pragma unroll
  for (int r = 0; r < 8; ++r) { p1[r] = 0.f; p2[r] = 0.f; }
#pragma unroll
  for (int t = 0; t < 4; ++t) {
    const float w1 = a1w[t * 16 + nn];
    const float w2 = a2w[t * 16 + nn];
#pragma unroll
    for (int r = 0; r < 8; ++r) {
      p1[r] += acc[t][r] * w1;
      p2[r] += acc[t][r] * w2;
    }
  }
#pragma unroll
  for (int mask = 1; mask < 16; mask <<= 1) {
#pragma unroll
    for (int r = 0; r < 8; ++r) {
      p1[r] += __shfl_xor(p1[r], mask, 32);
      p2[r] += __shfl_xor(p2[r], mask, 32);
    }
  }
  if (nn == 0) {
    const size_t base = (size_t)b * N + i0 + 8 * hi;
#pragma unroll
    for (int r = 0; r < 8; ++r) {
      f1[base + r] = p1[r] + a1b[0];
      f2[base + r] = p2[r] + a2b[0];
    }
  }
}

// ===========================================================================
// Kernel 2: one workgroup (8 waves) per 16-row query block.
// Each wave owns j-stripes of 32 (stride 256), online softmax, P@V via
// V_WMMA_F32_16X16X32_F16. A (P, 16x32 f16) built directly in WMMA lane
// layout: lane (hi,m): VGPR q holds K pair (idx 2q,2q+1),
//   j(idx) = j0 + (idx&7) + 8*(2*(idx>>3) + hi).
// B (V, 32x16 f16): lane (hi,n): VGPR v holds K pair (idx 2v,2v+1) at j(idx) as for A, loaded
// as two contiguous 16B chunks from transposed Vt.
// bias_mat is a one-pass 536 MB stream: non-temporal loads so it does not
// evict the L2-resident Vt; one-cacheline-per-row prefetch for the next tile.
// ===========================================================================
__global__ __launch_bounds__(256) void attn_kernel(
    const float* __restrict__ bias_mat, const _Float16* __restrict__ Vt,
    const float* __restrict__ f1g, const float* __restrict__ f2g,
    const float* __restrict__ obias, float* __restrict__ out, int N)
{
  __shared__ float ldsO[8][16][64];
  __shared__ float ldsM[8][16];
  __shared__ float ldsL[8][16];
  __shared__ float ldsS[8][16];
  __shared__ float ldsLinv[16];

  const int lane = threadIdx.x & 31;
  const int w    = threadIdx.x >> 5;
  const int blk  = blockIdx.x;     // B*N/16 = 1024 blocks
  const int b    = blk >> 9;
  const int i0   = (blk & 511) << 4;
  const int hi   = lane >> 4;
  const int m    = lane & 15;

  const float  f1row = f1g[(size_t)b * N + i0 + m];
  const float* brow  = bias_mat + ((size_t)b * N + i0 + m) * N;
  const float* f2b   = f2g + (size_t)b * N;
  const _Float16* Vb = Vt + (size_t)b * 64 * N;

  v8f acc[4];
#pragma unroll
  for (int t = 0; t < 4; ++t) acc[t] = (v8f)0.f;
  float mrun = -__builtin_inff();
  float lrun = 0.f;

  const int nchunks = N / 256;
  for (int it = 0; it < nchunks; ++it) {
    const int j0 = it * 256 + w * 32;

    // Prefetch next chunk's bias rows (one 128B line per row; lane pairs
    // share a line so one prefetch op covers the whole 16x32 tile).
    if (it + 1 < nchunks)
      __builtin_prefetch(brow + j0 + 256, 0, 0);

    const float f2v = f2b[j0 + lane];  // coalesced; redistributed via shfl

    float s[16];
    float cmax = -__builtin_inff();
#pragma unroll
    for (int g = 0; g < 4; ++g) {
      const int jg = (g & 1) * 4 + (g >> 1) * 16 + 8 * hi;
      // Streamed exactly once -> non-temporal.
      const v4f bm = __builtin_nontemporal_load((const v4f*)(brow + j0 + jg));
#pragma unroll
      for (int q = 0; q < 4; ++q) {
        const int idx = g * 4 + q;
        const float f2k = __shfl(f2v, jg + q, 32);
        float sc = f1row + f2k;
        sc = (sc >= 0.f) ? sc : 0.01f * sc;   // leaky relu
        sc += bm[q];                           // + bias_mat
        s[idx] = sc;
        cmax = fmaxf(cmax, sc);
      }
    }

    // Row max is shared by lane pair (lane, lane^16).
    cmax = fmaxf(cmax, __shfl_xor(cmax, 16, 32));
    const float mnew    = fmaxf(mrun, cmax);
    const float rescale = fexp2((mrun - mnew) * LOG2E);
    mrun = mnew;

    v16h av;
    float psum = 0.f;
#pragma unroll
    for (int idx = 0; idx < 16; ++idx) {
      const float p = fexp2((s[idx] - mnew) * LOG2E);
      psum += p;
      av[idx] = (_Float16)p;
    }
    psum += __shfl_xor(psum, 16, 32);
    lrun = lrun * rescale + psum;

    // Rescale accumulators: C-layout row for VGPR r is r + 8*hi; the scale
    // for row x lives in lane x (rows duplicated across halves).
    float rs[8];
#pragma unroll
    for (int r = 0; r < 8; ++r) rs[r] = __shfl(rescale, r + 8 * hi, 32);
#pragma unroll
    for (int t = 0; t < 4; ++t) {
#pragma unroll
      for (int r = 0; r < 8; ++r) acc[t][r] *= rs[r];
    }

    // P @ V for the 4 feature tiles (Vt is small and L2-hot).
#pragma unroll
    for (int t = 0; t < 4; ++t) {
      const _Float16* vp = Vb + (size_t)(t * 16 + m) * N + j0 + 8 * hi;
      union { v16h v; v8h h[2]; } bu;
      bu.h[0] = *(const v8h*)(vp);
      bu.h[1] = *(const v8h*)(vp + 16);
      acc[t] = __builtin_amdgcn_wmma_f32_16x16x32_f16(
          false, av, false, bu.v, (short)0, acc[t], false, false);
    }
  }

  // ---- merge the 8 per-wave partial softmaxes through LDS ----
  if (lane < 16) {
    ldsM[w][lane] = mrun;
    ldsL[w][lane] = lrun;
  }
#pragma unroll
  for (int t = 0; t < 4; ++t) {
#pragma unroll
    for (int r = 0; r < 8; ++r)
      ldsO[w][r + 8 * hi][t * 16 + m] = acc[t][r];
  }
  __syncthreads();

  if (threadIdx.x < 16) {
    const int row = threadIdx.x;
    float M = -__builtin_inff();
    for (int ww = 0; ww < 8; ++ww) M = fmaxf(M, ldsM[ww][row]);
    float L = 0.f;
    for (int ww = 0; ww < 8; ++ww) {
      const float sc = fexp2((ldsM[ww][row] - M) * LOG2E);
      ldsS[ww][row] = sc;
      L += ldsL[ww][row] * sc;
    }
    ldsLinv[row] = 1.0f / L;
  }
  __syncthreads();

  const size_t obase = ((size_t)b * N + i0) * 64;
#pragma unroll
  for (int q = 0; q < 4; ++q) {
    const int e   = threadIdx.x + q * 256;  // 16*64 = 1024 outputs per block
    const int row = e >> 6;
    const int f   = e & 63;
    float o = 0.f;
#pragma unroll
    for (int ww = 0; ww < 8; ++ww) o += ldsO[ww][row][f] * ldsS[ww][row];
    // Written once -> non-temporal store.
    __builtin_nontemporal_store(o * ldsLinv[row] + obias[f], out + obase + e);
  }
}

// ===========================================================================
extern "C" void kernel_launch(void* const* d_in, const int* in_sizes, int n_in,
                              void* d_out, int out_size, void* d_ws, size_t ws_size,
                              hipStream_t stream) {
  (void)in_sizes; (void)n_in; (void)out_size; (void)ws_size;
  const int B = 2, N = 8192;

  const float* seq      = (const float*)d_in[0];
  const float* bias_mat = (const float*)d_in[1];
  const float* W_w      = (const float*)d_in[2];
  const float* a1_w     = (const float*)d_in[3];
  const float* a1_b     = (const float*)d_in[4];
  const float* a2_w     = (const float*)d_in[5];
  const float* a2_b     = (const float*)d_in[6];
  const float* obias    = (const float*)d_in[7];

  char* ws = (char*)d_ws;
  _Float16* Vt = (_Float16*)ws;                                   // B*64*N f16 = 2 MB
  float* f1 = (float*)(ws + (size_t)B * 64 * N * sizeof(_Float16));
  float* f2 = f1 + (size_t)B * N;

  // K1: 1024 tiles, 8 waves per block -> 128 blocks.
  proj_kernel<<<(B * N / 16) / 8, 256, 0, stream>>>(
      seq, W_w, a1_w, a1_b, a2_w, a2_b, Vt, f1, f2, N);
  // K2: one block per 16-row query block.
  attn_kernel<<<B * (N / 16), 256, 0, stream>>>(
      bias_mat, Vt, f1, f2, obias, (float*)d_out, N);
}
